// PerlinAttention_67018669687424
// MI455X (gfx1250) — compile-verified, matrix operand order checked by reading, not run on hardware
//
#include <hip/hip_runtime.h>
#include <hip/hip_bf16.h>
#include <math.h>

// ---------------------------------------------------------------------------
// Perlin attention (Performer features + chunked causal linear attention +
// attention predictor) for MI455X / gfx1250.  All GEMM-shaped contractions go
// through v_wmma_f32_16x16x32_bf16 (wave32 WMMA, f32 accumulate).
// ---------------------------------------------------------------------------

typedef __attribute__((ext_vector_type(16))) __bf16 v16bf;
typedef __attribute__((ext_vector_type(8)))  __bf16 v8bf;
typedef __attribute__((ext_vector_type(8)))  float  v8f;
typedef __attribute__((ext_vector_type(4)))  float  v4f;

static constexpr int kB  = 2;
static constexpr int kH  = 12;
static constexpr int kBH = kB * kH;          // 24
static constexpr int kN  = 2048;
static constexpr int kD  = 64;
static constexpr int kM  = 266;              // performer features
static constexpr int kMP = 272;              // padded to 17*16
static constexpr int kE  = 192;              // value hidden (3*D)
static constexpr int kES = 48;               // E-slice per attention workgroup
static constexpr int kC2 = 128;              // predictor hidden (2*D)
static constexpr int kT  = 128;              // predictor length
static constexpr int kCK = 128;              // chunk
static constexpr int kNC = kN / kCK;         // 16 chunks

static constexpr float KERNEL_EPS = 1e-3f;
static constexpr float D_EPS      = 1e-6f;
static constexpr float LN_EPS     = 1e-5f;

// Native bf16 converts (gfx1250 has hardware bf16 cvt; RNE matches reference).
__device__ __forceinline__ __bf16 f2bf(float f) { return (__bf16)f; }
__device__ __forceinline__ float  bf2f(__bf16 b) { return (float)b; }

// v_wmma_f32_16x16x32_bf16 wrapper: D = A(16x32) * B(32x16) + C
__device__ __forceinline__ v8f wmma_bf(v16bf a, v16bf b, v8f c) {
  return __builtin_amdgcn_wmma_f32_16x16x32_bf16(false, a, false, b,
                                                 (short)0, c, false, false);
}

// Fragment loader from a row-major [row][k] bf16 buffer (global or LDS).
// ISA 16-bit operand layout: lane half h holds K = {k0+8h .. k0+8h+7} in the
// low 8 elements and K = {k0+16+8h ..} in the high 8 => two b128 loads.
__device__ __forceinline__ v16bf frag_rm(const __bf16* p, int k0, int half,
                                         int kvalid) {
  int lo = k0 + 8 * half;
  int hi = lo + 16;
  v8bf tlo, thi;
  if (lo + 8 <= kvalid) {
    tlo = *(const v8bf*)(p + lo);
  } else {
#pragma unroll
    for (int j = 0; j < 8; ++j) tlo[j] = (lo + j < kvalid) ? p[lo + j] : (__bf16)0.0f;
  }
  if (hi + 8 <= kvalid) {
    thi = *(const v8bf*)(p + hi);
  } else {
#pragma unroll
    for (int j = 0; j < 8; ++j) thi[j] = (hi + j < kvalid) ? p[hi + j] : (__bf16)0.0f;
  }
  return __builtin_shufflevector(tlo, thi, 0, 1, 2, 3, 4, 5, 6, 7,
                                 8, 9, 10, 11, 12, 13, 14, 15);
}

// Same layout, but from an fp32 row-major source with on-the-fly scale+cvt.
__device__ __forceinline__ v16bf frag_rm_f32(const float* p, int k0, int half,
                                             float scale) {
  int lo = k0 + 8 * half;
  int hi = lo + 16;
  v4f a0 = *(const v4f*)(p + lo);
  v4f a1 = *(const v4f*)(p + lo + 4);
  v4f b0 = *(const v4f*)(p + hi);
  v4f b1 = *(const v4f*)(p + hi + 4);
  v16bf f;
#pragma unroll
  for (int j = 0; j < 4; ++j) {
    f[j]      = (__bf16)(a0[j] * scale);
    f[4 + j]  = (__bf16)(a1[j] * scale);
    f[8 + j]  = (__bf16)(b0[j] * scale);
    f[12 + j] = (__bf16)(b1[j] * scale);
  }
  return f;
}

// ---------------------------------------------------------------------------
// Kernel 1: generalized features  qf/kf = relu((x/d^.25) @ proj^T) + eps
// grid: (BH*N/128, 2)   block: 256 (8 waves, wave w owns rows 16w..16w+15)
// LDS: proj staged bf16 [272][72]  (rows >= 266 zeroed)
// ---------------------------------------------------------------------------
__global__ void k_features(const float* __restrict__ q,
                           const float* __restrict__ k,
                           const float* __restrict__ proj,
                           __bf16* __restrict__ qf, __bf16* __restrict__ kf) {
  extern __shared__ char smem[];
  __bf16* pl = (__bf16*)smem;                       // [272][72]
  const float* x = blockIdx.y ? k : q;
  __bf16* o = blockIdx.y ? kf : qf;
  const int row0 = blockIdx.x * 128;
  const int t = threadIdx.x;
  for (int idx = t; idx < kMP * kD; idx += 256) {
    int m = idx >> 6, d = idx & 63;
    float v = (m < kM) ? proj[m * kD + d] : 0.f;
    pl[m * 72 + d] = f2bf(v);
  }
  __syncthreads();

  const int w = t >> 5, lane = t & 31, half = lane >> 4, r = lane & 15;
  const float dn = 0.35355339059327373f;            // 64^-0.25
  const float* xrow = x + (long)(row0 + w * 16 + r) * kD;
  v16bf a0 = frag_rm_f32(xrow, 0, half, dn);
  v16bf a1 = frag_rm_f32(xrow, 32, half, dn);

  for (int m0 = 0; m0 < 17; ++m0) {
    v8f acc = {};
    const __bf16* bp = pl + (m0 * 16 + r) * 72;
    acc = wmma_bf(a0, frag_rm(bp, 0, half, 64), acc);
    acc = wmma_bf(a1, frag_rm(bp, 32, half, 64), acc);
    int col = m0 * 16 + r;
#pragma unroll
    for (int v = 0; v < 8; ++v) {
      int row = row0 + w * 16 + v + 8 * half;
      float val = (col < kM) ? (fmaxf(acc[v], 0.f) + KERNEL_EPS) : 0.f;
      o[(long)row * kMP + col] = f2bf(val);
    }
  }
}

// ---------------------------------------------------------------------------
// Kernel 2: chunked causal linear attention.
// grid: (24 bh, 4 e-slices)  block: 256.  Sequential 16-chunk scan per wg.
// LDS (~150KB): S^T bf16 [48][280], kc^T bf16 [272][136], v^T bf16 [48][136],
//               qk bf16 [128][136], ksum f32[272], dbuf f32[128]
// ---------------------------------------------------------------------------
__global__ void k_attn(const __bf16* __restrict__ qf,
                       const __bf16* __restrict__ kf,
                       const float* __restrict__ v3,
                       float* __restrict__ ctx) {
  extern __shared__ char smem[];
  __bf16* ST  = (__bf16*)smem;                      // [48][280]  S^T slice
  __bf16* KCM = ST + kES * 280;                     // [272][136] kc^T
  __bf16* VCT = KCM + kMP * 136;                    // [48][136]  vc^T slice
  __bf16* QK  = VCT + kES * 136;                    // [128][136] masked qk
  float*  ksum = (float*)(QK + kCK * 136);          // [272]
  float*  dbuf = ksum + kMP;                        // [128]

  const int bh   = blockIdx.x;
  const int eblk = blockIdx.y;                      // e-slice: eblk*48
  const int e0g  = eblk * kES;
  const __bf16* qfb = qf + (long)bh * kN * kMP;
  const __bf16* kfb = kf + (long)bh * kN * kMP;
  const float*  vb  = v3 + (long)bh * kN * kE;
  float*        cb  = ctx + (long)bh * kN * kE;

  const int t = threadIdx.x, w = t >> 5, lane = t & 31;
  const int half = lane >> 4, r = lane & 15;

  for (int i = t; i < kES * 280; i += 256) ST[i] = (__bf16)0.0f;
  for (int i = t; i < kMP; i += 256) ksum[i] = 0.f;
  __syncthreads();

  for (int c = 0; c < kNC; ++c) {
    const __bf16* qc = qfb + (long)c * kCK * kMP;
    const __bf16* kc = kfb + (long)c * kCK * kMP;
    const float*  vc = vb + (long)c * kCK * kE;

    // Prefetch next chunk's streams into cache (global_prefetch_b8).
    if (c + 1 < kNC) {
      const __bf16* qn = qc + (long)kCK * kMP;
      const __bf16* kn = kc + (long)kCK * kMP;
      const float*  vn = vc + (long)kCK * kE;
      __builtin_prefetch(qn + t * 64, 0, 3);
      __builtin_prefetch(kn + t * 64, 0, 3);
      __builtin_prefetch(vn + t * 64, 0, 3);
    }

    // ---- phase 0: stage kc^T, vc^T; init denominator with old ksum ----
    for (int idx = t; idx < kCK * kMP; idx += 256) {
      int cc = idx / kMP, m = idx % kMP;            // coalesced read
      KCM[m * 136 + cc] = kc[(long)cc * kMP + m];
    }
    for (int idx = t; idx < kCK * kES; idx += 256) {
      int cc = idx / kES, e = idx % kES;
      VCT[e * 136 + cc] = f2bf(vc[(long)cc * kE + e0g + e]);
    }
    if (t < kCK) {
      const __bf16* qr = qc + (long)t * kMP;
      float s = 0.f;
      for (int m = 0; m < kMP; m += 8) {
        v8bf qv = *(const v8bf*)(qr + m);
#pragma unroll
        for (int j = 0; j < 8; ++j) s += bf2f(qv[j]) * (ksum[m + j] + D_EPS);
      }
      dbuf[t] = s;
    }
    __syncthreads();

    // ---- phase 1: qk = tril(qc @ kc^T); wave w owns row strip i0 = 16w ----
    v16bf aq[9];
    const __bf16* qrow = qc + (long)(w * 16 + r) * kMP;
#pragma unroll
    for (int ks = 0; ks < 9; ++ks) aq[ks] = frag_rm(qrow, ks * 32, half, kMP);

    float rsum[8];
#pragma unroll
    for (int v = 0; v < 8; ++v) rsum[v] = 0.f;

    for (int j0 = 0; j0 < 8; ++j0) {
      if (j0 <= w) {
        v8f acc = {};
        const __bf16* krow = kc + (long)(j0 * 16 + r) * kMP;
#pragma unroll
        for (int ks = 0; ks < 9; ++ks)
          acc = wmma_bf(aq[ks], frag_rm(krow, ks * 32, half, kMP), acc);
#pragma unroll
        for (int v = 0; v < 8; ++v) {
          int i = w * 16 + v + 8 * half, j = j0 * 16 + r;
          float x = (j <= i) ? acc[v] : 0.f;
          rsum[v] += x;
          QK[i * 136 + j] = f2bf(x);
        }
      } else {
#pragma unroll
        for (int v = 0; v < 8; ++v)
          QK[(w * 16 + v + 8 * half) * 136 + j0 * 16 + r] = (__bf16)0.0f;
      }
    }
#pragma unroll
    for (int v = 0; v < 8; ++v)
      atomicAdd(&dbuf[w * 16 + v + 8 * half], rsum[v]);
    __syncthreads();

    // ---- phase 2: out = (qc @ S_old + qk @ vc) / d ----
    v16bf aqk[4];
    const __bf16* qkrow = QK + (w * 16 + r) * 136;
#pragma unroll
    for (int ks = 0; ks < 4; ++ks) aqk[ks] = frag_rm(qkrow, ks * 32, half, kCK);

    for (int e0 = 0; e0 < kES / 16; ++e0) {
      v8f acc = {};
      const __bf16* srow = ST + (e0 * 16 + r) * 280;
#pragma unroll
      for (int ks = 0; ks < 9; ++ks)
        acc = wmma_bf(aq[ks], frag_rm(srow, ks * 32, half, kMP), acc);
      const __bf16* vrow = VCT + (e0 * 16 + r) * 136;
#pragma unroll
      for (int ks = 0; ks < 4; ++ks)
        acc = wmma_bf(aqk[ks], frag_rm(vrow, ks * 32, half, kCK), acc);
#pragma unroll
      for (int v = 0; v < 8; ++v) {
        int i = w * 16 + v + 8 * half;
        cb[(long)(c * kCK + i) * kE + e0g + e0 * 16 + r] = acc[v] / dbuf[i];
      }
    }
    __syncthreads();

    // ---- phase 3: ksum += colsum(kc);  S^T += vc^T @ kc ----
    for (int m = t; m < kMP; m += 256) {
      float s = ksum[m];
      for (int cc = 0; cc < kCK; ++cc) s += bf2f(KCM[m * 136 + cc]);
      ksum[m] = s;
    }
    for (int tile = w; tile < (kES / 16) * 17; tile += 8) {
      int e0 = tile / 17, m0 = tile % 17;
      v8f acc;
#pragma unroll
      for (int v = 0; v < 8; ++v)
        acc[v] = bf2f(ST[(e0 * 16 + v + 8 * half) * 280 + m0 * 16 + r]);
      const __bf16* arow = VCT + (e0 * 16 + r) * 136;
      const __bf16* brow = KCM + (m0 * 16 + r) * 136;
#pragma unroll
      for (int ks = 0; ks < 4; ++ks)
        acc = wmma_bf(frag_rm(arow, ks * 32, half, kCK),
                      frag_rm(brow, ks * 32, half, kCK), acc);
#pragma unroll
      for (int v = 0; v < 8; ++v)
        ST[(e0 * 16 + v + 8 * half) * 280 + m0 * 16 + r] = f2bf(acc[v]);
    }
    __syncthreads();
  }
}

// ---------------------------------------------------------------------------
// Kernel 3: predictor  Linear(192->128) -> LN -> GELU -> Linear(128->128)
//           -> softmax.   grid: BH*N/128, block 256.
// ---------------------------------------------------------------------------
__global__ void k_pred(const float* __restrict__ ctx,
                       const float* __restrict__ enc_w,
                       const float* __restrict__ enc_b,
                       const float* __restrict__ ln_g,
                       const float* __restrict__ ln_b,
                       const float* __restrict__ dec_w,
                       const float* __restrict__ dec_b,
                       float* __restrict__ probs) {
  extern __shared__ char smem[];
  __bf16* EW = (__bf16*)smem;                       // [128][200] enc_w^T
  __bf16* DW = EW + kC2 * 200;                      // [128][136] dec_w^T
  __bf16* HB = DW + kT * 136;                       // [128][136] gelu(h) bf16
  float*  HS = (float*)(HB + kCK * 136);            // [128][136] f32 scratch

  const long row0 = (long)blockIdx.x * 128;
  const int t = threadIdx.x, w = t >> 5, lane = t & 31;
  const int half = lane >> 4, r = lane & 15;

  for (int idx = t; idx < kE * kC2; idx += 256) {   // enc_w [192][128] -> [c][e]
    int e = idx / kC2, cc = idx % kC2;
    EW[cc * 200 + e] = f2bf(enc_w[idx]);
  }
  for (int idx = t; idx < kC2 * kT; idx += 256) {   // dec_w [128][128] -> [t][c]
    int cc = idx / kT, tt = idx % kT;
    DW[tt * 136 + cc] = f2bf(dec_w[idx]);
  }
  __syncthreads();

  {                                                 // enc GEMM
    v16bf a[6];
    const float* xrow = ctx + (row0 + w * 16 + r) * kE;
#pragma unroll
    for (int ks = 0; ks < 6; ++ks) a[ks] = frag_rm_f32(xrow, ks * 32, half, 1.f);
    for (int c0 = 0; c0 < 8; ++c0) {
      v8f acc = {};
      const __bf16* brow = EW + (c0 * 16 + r) * 200;
#pragma unroll
      for (int ks = 0; ks < 6; ++ks)
        acc = wmma_bf(a[ks], frag_rm(brow, ks * 32, half, kE), acc);
      float bias = enc_b[c0 * 16 + r];
#pragma unroll
      for (int v = 0; v < 8; ++v)
        HS[(w * 16 + v + 8 * half) * 136 + c0 * 16 + r] = acc[v] + bias;
    }
  }
  __syncthreads();

  if (t < 128) {                                    // LayerNorm + exact GELU
    float* h = HS + t * 136;
    float mu = 0.f;
    for (int cc = 0; cc < kC2; ++cc) mu += h[cc];
    mu *= (1.f / kC2);
    float var = 0.f;
    for (int cc = 0; cc < kC2; ++cc) { float d = h[cc] - mu; var += d * d; }
    var *= (1.f / kC2);
    float rs = rsqrtf(var + LN_EPS);
    for (int cc = 0; cc < kC2; ++cc) {
      float y = (h[cc] - mu) * rs * ln_g[cc] + ln_b[cc];
      y = 0.5f * y * (1.f + erff(y * 0.70710678118654752f));
      HB[t * 136 + cc] = f2bf(y);
    }
  }
  __syncthreads();

  {                                                 // dec GEMM
    v16bf a[4];
    const __bf16* hrow = HB + (w * 16 + r) * 136;
#pragma unroll
    for (int ks = 0; ks < 4; ++ks) a[ks] = frag_rm(hrow, ks * 32, half, kC2);
    for (int t0 = 0; t0 < 8; ++t0) {
      v8f acc = {};
      const __bf16* brow = DW + (t0 * 16 + r) * 136;
#pragma unroll
      for (int ks = 0; ks < 4; ++ks)
        acc = wmma_bf(a[ks], frag_rm(brow, ks * 32, half, kC2), acc);
      float bias = dec_b[t0 * 16 + r];
#pragma unroll
      for (int v = 0; v < 8; ++v)
        HS[(w * 16 + v + 8 * half) * 136 + t0 * 16 + r] = acc[v] + bias;
    }
  }
  __syncthreads();

  if (t < 128) {                                    // softmax
    float* s = HS + t * 136;
    float mx = -INFINITY;
    for (int tt = 0; tt < kT; ++tt) mx = fmaxf(mx, s[tt]);
    float sum = 0.f;
    for (int tt = 0; tt < kT; ++tt) { float e = expf(s[tt] - mx); s[tt] = e; sum += e; }
    float inv = 1.f / sum;
    float* o = probs + (row0 + t) * kT;
    for (int tt = 0; tt < kT; ++tt) o[tt] = s[tt] * inv;
  }
}

// ---------------------------------------------------------------------------
extern "C" void kernel_launch(void* const* d_in, const int* in_sizes, int n_in,
                              void* d_out, int out_size, void* d_ws,
                              size_t ws_size, hipStream_t stream) {
  const float* q     = (const float*)d_in[0];
  const float* k     = (const float*)d_in[1];
  const float* v3    = (const float*)d_in[2];
  const float* proj  = (const float*)d_in[3];
  const float* enc_w = (const float*)d_in[4];
  const float* enc_b = (const float*)d_in[5];
  const float* ln_g  = (const float*)d_in[6];
  const float* ln_b  = (const float*)d_in[7];
  const float* dec_w = (const float*)d_in[8];
  const float* dec_b = (const float*)d_in[9];

  float* ctx   = (float*)d_out;                        // [24,2048,192]
  float* probs = ctx + (long)kBH * kN * kE;            // [24,2048,128]

  // workspace: bf16 qf/kf features [24][2048][272]  (2 * 26.7 MB)
  __bf16* qf = (__bf16*)d_ws;
  __bf16* kf = qf + (long)kBH * kN * kMP;

  const size_t smem1 = (size_t)kMP * 72 * 2;                              // 39168
  const size_t smem2 = (size_t)(kES * 280 + kMP * 136 + kES * 136 +
                                kCK * 136) * 2 + (kMP + kCK) * 4;         // ~150KB
  const size_t smem3 = (size_t)(kC2 * 200 + kT * 136 + kCK * 136) * 2 +
                       (size_t)kCK * 136 * 4;                             // ~190KB

  dim3 g1(kBH * kN / 128, 2);
  k_features<<<g1, 256, smem1, stream>>>(q, k, proj, qf, kf);

  dim3 g2(kBH, kE / kES);                              // 24 x 4 = 96 wgs
  k_attn<<<g2, 256, smem2, stream>>>(qf, kf, v3, ctx);

  k_pred<<<kBH * kN / 128, 256, smem3, stream>>>(ctx, enc_w, enc_b, ln_g, ln_b,
                                                 dec_w, dec_b, probs);
}
